// GAT_HE_19344532701816
// MI455X (gfx1250) — matrix operand order repaired and checked by reading, not compiled, not run on hardware
//
#include <hip/hip_runtime.h>
#include <hip/hip_bf16.h>

#define NNODES 100000
#define NEDGES 500000
#define DD 128
#define HH 3
#define NEGS 0.2f

typedef __attribute__((ext_vector_type(16))) _Float16 v16h;
typedef __attribute__((ext_vector_type(8)))  float    v8f;

// ---------------------------------------------------------------------------
// Wc[h] = W_lin @ W_head[h]   (3 x 128x128x128, tiny)
// Also emits WcH: f16 copy pre-swizzled into the WMMA B-fragment LDS image:
// fragment (ks,nt): lane holds col = n&15 (+16 if k&16), element i = k&15.
// ---------------------------------------------------------------------------
__global__ void k_wc(const float* __restrict__ Wlin,
                     const float* __restrict__ Whead,
                     float* __restrict__ Wc,
                     _Float16* __restrict__ WcH) {
    int idx = blockIdx.x * blockDim.x + threadIdx.x;   // over 3*128*128
    if (idx >= HH * DD * DD) return;
    int h = idx / (DD * DD);
    int r = idx - h * DD * DD;
    int k = r / DD, n = r - (r / DD) * DD;
    const float* wl = Wlin + k * DD;
    const float* wh = Whead + h * DD * DD + n;
    float acc = 0.f;
#pragma unroll 8
    for (int d = 0; d < DD; ++d) acc += wl[d] * wh[d * DD];
    Wc[idx] = acc;

    // swizzled f16 image (same layout the GEMM kernel wants in LDS)
    int ks = k >> 5, kin = k & 31, nt = n >> 4, nin = n & 15;
    int lane2 = nin + ((kin & 16) ? 16 : 0);
    int i = kin & 15;
    WcH[(size_t)h * DD * DD + ((((ks << 3) + nt) << 5) + lane2) * 16 + i] = (_Float16)acc;
}

// ---------------------------------------------------------------------------
// v_src[h] = Wc[h] @ att_src[h]  (and same for dst) so that
// a_src[h,n] = x[n] . v_src[h]
// ---------------------------------------------------------------------------
__global__ void k_v(const float* __restrict__ Wc,
                    const float* __restrict__ attS,
                    const float* __restrict__ attD,
                    float* __restrict__ vs, float* __restrict__ vd) {
    int idx = blockIdx.x * blockDim.x + threadIdx.x;   // over 2*3*128
    if (idx >= 2 * HH * DD) return;
    int which = idx >= HH * DD;
    int j = idx - which * HH * DD;
    int h = j / DD, k = j - h * DD;
    const float* a = (which ? attD : attS) + h * DD;
    const float* w = Wc + ((size_t)h * DD + k) * DD;
    float acc = 0.f;
#pragma unroll 8
    for (int d = 0; d < DD; ++d) acc += w[d] * a[d];
    (which ? vd : vs)[j] = acc;
}

// ---------------------------------------------------------------------------
// xh[h] = x @ Wc[h]  via v_wmma_f32_16x16x32_f16.
// Block = 256 thr (8 waves) -> 128 rows of one head.
// B image (pre-swizzled f16, 32 KB) is DMA'd into LDS with CDNA5
// global_load_async_to_lds_b128 (ASYNCcnt), then waves run 8x4 WMMAs.
// A fragments built from global f32 x with on-the-fly f16 convert, following
// a contiguous k order, matching the B image (lane<16: K 0..15 ; lane>=16: K 16..31).
// ---------------------------------------------------------------------------
__global__ __launch_bounds__(256)
void k_gemm(const float* __restrict__ x,
            const _Float16* __restrict__ WcH,
            float* __restrict__ XH) {
    __shared__ _Float16 Bs[DD * DD];   // 32 KB

    const int nbRow = (NNODES + 127) / 128;
    int h  = blockIdx.x / nbRow;
    int rb = blockIdx.x - h * nbRow;
    int tid = threadIdx.x;

    // ---- async DMA of the pre-swizzled B image into LDS ----
    {
        unsigned ldsBase = (unsigned)(uintptr_t)(&Bs[0]);   // low 32b of flat = LDS offset
        unsigned long long g = (unsigned long long)(uintptr_t)(WcH + (size_t)h * DD * DD);
        unsigned loff = ldsBase + tid * 16;
        unsigned long long gaddr = g + tid * 16;
#pragma unroll
        for (int i = 0; i < 8; ++i) {      // 256 thr * 16 B * 8 = 32 KB
            asm volatile("global_load_async_to_lds_b128 %0, %1, off"
                         :: "v"(loff), "v"(gaddr) : "memory");
            loff  += 4096;
            gaddr += 4096;
        }
        asm volatile("s_wait_asynccnt 0x0" ::: "memory");
    }
    __syncthreads();

    int wave = tid >> 5, lane = tid & 31;
    int rowBase = rb * 128 + wave * 16;
    int r = lane & 15;
    int khalf = (lane >> 4) << 4;            // 0 or 16
    int gr = rowBase + r;
    if (gr > NNODES - 1) gr = NNODES - 1;    // clamp (keeps EXEC uniform)
    const float* xr = x + (size_t)gr * DD;

    // ---- A fragments for 4 K-steps ----
    v16h A[4];
#pragma unroll
    for (int ks = 0; ks < 4; ++ks) {
        int k0 = ks * 32 + khalf;
        float4 p0 = *(const float4*)(xr + k0);
        float4 p1 = *(const float4*)(xr + k0 + 4);
        float4 p2 = *(const float4*)(xr + k0 + 8);
        float4 p3 = *(const float4*)(xr + k0 + 12);
        A[ks][0]  = (_Float16)p0.x; A[ks][1]  = (_Float16)p0.y;
        A[ks][2]  = (_Float16)p0.z; A[ks][3]  = (_Float16)p0.w;
        A[ks][4]  = (_Float16)p1.x; A[ks][5]  = (_Float16)p1.y;
        A[ks][6]  = (_Float16)p1.z; A[ks][7]  = (_Float16)p1.w;
        A[ks][8]  = (_Float16)p2.x; A[ks][9]  = (_Float16)p2.y;
        A[ks][10] = (_Float16)p2.z; A[ks][11] = (_Float16)p2.w;
        A[ks][12] = (_Float16)p3.x; A[ks][13] = (_Float16)p3.y;
        A[ks][14] = (_Float16)p3.z; A[ks][15] = (_Float16)p3.w;
    }

    float* outBase = XH + (size_t)h * NNODES * DD;
    int col0 = lane & 15;
    int rOff = (lane >> 4) << 3;
    // NNODES % 16 == 0 -> a 16-row tile is entirely valid or entirely OOB
    bool tileValid = (rowBase < NNODES);

#pragma unroll
    for (int nt = 0; nt < 8; ++nt) {
        v8f acc = {};
#pragma unroll
        for (int ks = 0; ks < 4; ++ks) {
            const v16h b = *(const v16h*)(&Bs[((((ks << 3) + nt) << 5) + lane) * 16]);
            acc = __builtin_amdgcn_wmma_f32_16x16x32_f16(
                false, A[ks], false, b, (short)0, acc, false, false);
        }
        if (tileValid) {
            int col = nt * 16 + col0;
            float* o = outBase + (size_t)(rowBase + rOff) * DD + col;
#pragma unroll
            for (int rr = 0; rr < 8; ++rr) o[(size_t)rr * DD] = acc[rr];
        }
    }
}

// ---------------------------------------------------------------------------
// a_src[h,n] = x[n].v_src[h] ; a_dst likewise (v vectors cached in LDS)
// ---------------------------------------------------------------------------
__global__ __launch_bounds__(256)
void k_av(const float* __restrict__ x,
          const float* __restrict__ vs, const float* __restrict__ vd,
          float* __restrict__ aS, float* __restrict__ aD) {
    __shared__ float Vs[HH * DD], Vd[HH * DD];
    int tid = threadIdx.x;
    for (int i = tid; i < HH * DD; i += 256) { Vs[i] = vs[i]; Vd[i] = vd[i]; }
    __syncthreads();
    int n = blockIdx.x * 256 + tid;
    if (n >= NNODES) return;
    const float4* xr = (const float4*)(x + (size_t)n * DD);
    float s0 = 0, s1 = 0, s2 = 0, d0 = 0, d1 = 0, d2 = 0;
#pragma unroll 4
    for (int k4 = 0; k4 < DD / 4; ++k4) {
        float4 xv = xr[k4];
        int k = k4 * 4;
        float4 a0 = *(const float4*)&Vs[0 * DD + k];
        float4 a1 = *(const float4*)&Vs[1 * DD + k];
        float4 a2 = *(const float4*)&Vs[2 * DD + k];
        float4 b0 = *(const float4*)&Vd[0 * DD + k];
        float4 b1 = *(const float4*)&Vd[1 * DD + k];
        float4 b2 = *(const float4*)&Vd[2 * DD + k];
        s0 += xv.x*a0.x + xv.y*a0.y + xv.z*a0.z + xv.w*a0.w;
        s1 += xv.x*a1.x + xv.y*a1.y + xv.z*a1.z + xv.w*a1.w;
        s2 += xv.x*a2.x + xv.y*a2.y + xv.z*a2.z + xv.w*a2.w;
        d0 += xv.x*b0.x + xv.y*b0.y + xv.z*b0.z + xv.w*b0.w;
        d1 += xv.x*b1.x + xv.y*b1.y + xv.z*b1.z + xv.w*b1.w;
        d2 += xv.x*b2.x + xv.y*b2.y + xv.z*b2.z + xv.w*b2.w;
    }
    aS[0 * NNODES + n] = s0; aS[1 * NNODES + n] = s1; aS[2 * NNODES + n] = s2;
    aD[0 * NNODES + n] = d0; aD[1 * NNODES + n] = d1; aD[2 * NNODES + n] = d2;
}

// ---------------------------------------------------------------------------
// init: out = mean_h(bias[h]) ; m = -inf ; s = 0
// ---------------------------------------------------------------------------
__global__ void k_init(const float* __restrict__ bias,
                       float* __restrict__ out,
                       float* __restrict__ m, float* __restrict__ sg) {
    long long i = (long long)blockIdx.x * 256 + threadIdx.x;
    if (i < (long long)NNODES * DD) {
        int d = (int)(i & (DD - 1));
        out[i] = (bias[d] + bias[DD + d] + bias[2 * DD + d]) * (1.f / 3.f);
    }
    if (i < (long long)HH * NNODES) {
        ((unsigned*)m)[i] = 0xFF800000u;   // -inf
        sg[i] = 0.f;
    }
}

__device__ __forceinline__ float leaky(float z) { return z > 0.f ? z : NEGS * z; }

// ---------------------------------------------------------------------------
// pass 1 over edges: segment max of leaky-relu logits (float atomic max via
// sign-split int/uint atomics; memory initialized to -inf)
// ---------------------------------------------------------------------------
__global__ void k_e1(const int* __restrict__ ei,
                     const float* __restrict__ aS, const float* __restrict__ aD,
                     float* __restrict__ m) {
    int e = blockIdx.x * 256 + threadIdx.x;
    if (e >= NEDGES) return;
    int s = ei[e], d = ei[NEDGES + e];
#pragma unroll
    for (int h = 0; h < HH; ++h) {
        float z = leaky(aS[h * NNODES + s] + aD[h * NNODES + d]);
        float* addr = m + h * NNODES + d;
        if (z >= 0.f) atomicMax((int*)addr, __float_as_int(z));
        else          atomicMin((unsigned*)addr, __float_as_uint(z));
    }
}

// ---------------------------------------------------------------------------
// pass 2: segment sum of exp(logit - max)
// ---------------------------------------------------------------------------
__global__ void k_e2(const int* __restrict__ ei,
                     const float* __restrict__ aS, const float* __restrict__ aD,
                     const float* __restrict__ m, float* __restrict__ sg) {
    int e = blockIdx.x * 256 + threadIdx.x;
    if (e >= NEDGES) return;
    int s = ei[e], d = ei[NEDGES + e];
#pragma unroll
    for (int h = 0; h < HH; ++h) {
        float z = leaky(aS[h * NNODES + s] + aD[h * NNODES + d]);
        float ex = __expf(z - m[h * NNODES + d]);
        atomicAdd(&sg[h * NNODES + d], ex);
    }
}

// ---------------------------------------------------------------------------
// pass 3: one wave per edge. Lanes 0..2 compute per-head alpha (recomputed
// logits, all L2-hot), broadcast via shuffle; each lane handles 4 channels.
// Head-sum in registers first -> single atomic scatter per channel.
// ---------------------------------------------------------------------------
__global__ __launch_bounds__(256)
void k_e3(const int* __restrict__ ei, const int* __restrict__ ewi,
          const float* __restrict__ emb, const float* __restrict__ XH,
          const float* __restrict__ aS, const float* __restrict__ aD,
          const float* __restrict__ m, const float* __restrict__ sg,
          float* __restrict__ out) {
    int e = blockIdx.x * 8 + (threadIdx.x >> 5);
    if (e >= NEDGES) return;
    int lane = threadIdx.x & 31;
    int s = ei[e], d = ei[NEDGES + e];
    int w = ewi[e];

    float al = 0.f;
    if (lane < HH) {
        int h = lane;
        float z = leaky(aS[h * NNODES + s] + aD[h * NNODES + d]);
        float ex = __expf(z - m[h * NNODES + d]);
        al = ex / (sg[h * NNODES + d] + 1e-16f);
    }
    const float inv3 = 1.f / 3.f;
    float a0 = __shfl(al, 0) * inv3;
    float a1 = __shfl(al, 1) * inv3;
    float a2 = __shfl(al, 2) * inv3;

    int c = lane * 4;
    float4 e4 = *(const float4*)(emb + (size_t)w * DD + c);
    float4 x0 = *(const float4*)(XH + ((size_t)0 * NNODES + s) * DD + c);
    float4 x1 = *(const float4*)(XH + ((size_t)1 * NNODES + s) * DD + c);
    float4 x2 = *(const float4*)(XH + ((size_t)2 * NNODES + s) * DD + c);

    float4 acc;
    acc.x = (a0 * x0.x + a1 * x1.x + a2 * x2.x) * e4.x;
    acc.y = (a0 * x0.y + a1 * x1.y + a2 * x2.y) * e4.y;
    acc.z = (a0 * x0.z + a1 * x1.z + a2 * x2.z) * e4.z;
    acc.w = (a0 * x0.w + a1 * x1.w + a2 * x2.w) * e4.w;

    float* o = out + (size_t)d * DD + c;
    atomicAdd(o + 0, acc.x);
    atomicAdd(o + 1, acc.y);
    atomicAdd(o + 2, acc.z);
    atomicAdd(o + 3, acc.w);
}

// ---------------------------------------------------------------------------
extern "C" void kernel_launch(void* const* d_in, const int* in_sizes, int n_in,
                              void* d_out, int out_size, void* d_ws, size_t ws_size,
                              hipStream_t stream) {
    const float* x     = (const float*)d_in[0];
    const int*   ei    = (const int*)  d_in[1];   // [2, E]
    const int*   ewi   = (const int*)  d_in[2];   // [E]
    const float* Wlin  = (const float*)d_in[3];
    const float* emb   = (const float*)d_in[4];
    const float* Whead = (const float*)d_in[5];
    const float* attS  = (const float*)d_in[6];
    const float* attD  = (const float*)d_in[7];
    const float* bias  = (const float*)d_in[8];
    float* out = (float*)d_out;

    // workspace carve-up (floats): ~159 MB total
    float*    Wc  = (float*)d_ws;                       // 3*128*128 f32
    _Float16* WcH = (_Float16*)(Wc + HH * DD * DD);     // 3*128*128 f16 (swizzled)
    float*    vs  = Wc + HH * DD * DD + (HH * DD * DD) / 2;   // 3*128
    float*    vd  = vs + HH * DD;                       // 3*128
    float*    aS  = vd + HH * DD;                       // 3*N
    float*    aD  = aS + HH * NNODES;                   // 3*N
    float*    mM  = aD + HH * NNODES;                   // 3*N
    float*    sg  = mM + HH * NNODES;                   // 3*N
    float*    XH  = sg + HH * NNODES;                   // 3*N*128 (16B-aligned)

    k_wc<<<(HH * DD * DD + 255) / 256, 256, 0, stream>>>(Wlin, Whead, Wc, WcH);
    k_v<<<(2 * HH * DD + 255) / 256, 256, 0, stream>>>(Wc, attS, attD, vs, vd);

    int nbRow = (NNODES + 127) / 128;
    k_gemm<<<nbRow * HH, 256, 0, stream>>>(x, WcH, XH);

    k_av<<<(NNODES + 255) / 256, 256, 0, stream>>>(x, vs, vd, aS, aD);
    k_init<<<((long long)NNODES * DD + 255) / 256, 256, 0, stream>>>(bias, out, mM, sg);

    k_e1<<<(NEDGES + 255) / 256, 256, 0, stream>>>(ei, aS, aD, mM);
    k_e2<<<(NEDGES + 255) / 256, 256, 0, stream>>>(ei, aS, aD, mM, sg);
    k_e3<<<(NEDGES + 7) / 8, 256, 0, stream>>>(ei, ewi, emb, XH, aS, aD, mM, sg, out);
}
